// reinforce_cons_loss_9526237462535
// MI455X (gfx1250) — matrix operand order repaired and checked by reading, not compiled, not run on hardware
//
#include <hip/hip_runtime.h>
#include <stdint.h>

typedef _Float16 v16h __attribute__((ext_vector_type(16)));
typedef _Float16 v8h  __attribute__((ext_vector_type(8)));
typedef float    v8f  __attribute__((ext_vector_type(8)));

#define H 256
#define W 256
#define HW (H*W)
#define NS 10
#define NC 4

__device__ __forceinline__ uint64_t splitmix64(uint64_t x) {
    x += 0x9E3779B97F4A7C15ull;
    x = (x ^ (x >> 30)) * 0xBF58476D1CE4E5B9ull;
    x = (x ^ (x >> 27)) * 0x94D049BB133111EBull;
    return x ^ (x >> 31);
}

// ---------------------------------------------------------------------------
// K1: per-pixel categorical sampling (10 samples) + gather of sampled prob
// ---------------------------------------------------------------------------
__global__ void k_sample_gather(const float* __restrict__ prob,
                                uint8_t* __restrict__ smp,
                                float* __restrict__ psel, int nimg) {
    int idx = blockIdx.x * blockDim.x + threadIdx.x;   // over nimg*HW
    if (idx >= nimg * HW) return;
    int img = idx / HW;
    int pix = idx - img * HW;
    const float* p = prob + (size_t)img * NC * HW + pix;
    __builtin_prefetch(p + HW, 0, 0);                  // global_prefetch_b8
    float p0 = p[0], p1 = p[HW], p2 = p[2 * HW], p3 = p[3 * HW];
    float sum = p0 + p1 + p2 + p3;
    for (int s = 0; s < NS; s++) {
        uint64_t h = splitmix64(((uint64_t)(img * NS + s) << 32) ^
                                (uint64_t)(unsigned)pix ^ 0xA5A5F00D12345ull);
        float u = (float)((h >> 40) & 0xFFFFFF) * (1.0f / 16777216.0f);
        float t = u * sum;
        int c; float pc;
        if      (t < p0)           { c = 0; pc = p0; }
        else if (t < p0 + p1)      { c = 1; pc = p1; }
        else if (t < p0 + p1 + p2) { c = 2; pc = p2; }
        else                       { c = 3; pc = p3; }
        size_t o = (size_t)(img * NS + s) * HW + pix;
        smp[o]  = (uint8_t)c;
        psel[o] = pc;
    }
}

// ---------------------------------------------------------------------------
// K2: fg = (sample == cls) as f16
// ---------------------------------------------------------------------------
__global__ void k_fg(const uint8_t* __restrict__ smp, _Float16* __restrict__ fg,
                     int cls, int B) {
    int i = blockIdx.x * blockDim.x + threadIdx.x;
    if (i < B * HW) fg[i] = (smp[i] == cls) ? (_Float16)1.0f : (_Float16)0.0f;
}

// ---------------------------------------------------------------------------
// K3: dst = sumpool(src, 2R+1) * src  via banded WMMA GEMMs.
// out_tile(16x16) = A_band(16x48) * X_halo(48x48) * B_band(48x16)
// 4 waves per block, one 16x16 output tile per wave.
// LDS layouts are arranged so every WMMA operand gather is a contiguous,
// branch-free ds_load_b128 matching the documented wave32 fragment layouts.
// ---------------------------------------------------------------------------
template<int R>
__global__ void __launch_bounds__(128)
k_sumpool_mask_wmma(const _Float16* __restrict__ src, float* __restrict__ dst,
                    int B) {
    // haloT[x][y]: transposed halo, rows padded to 64 (rows 48..63 = 0)
    __shared__ __attribute__((aligned(32))) _Float16 haloT[4][48 * 64]; // 24,576 B
    // T spill: row-major 16x64 (cols 48..63 = 0)
    __shared__ __attribute__((aligned(32))) _Float16 tls[4][16 * 64];   //  8,192 B
    // band matrix: BND[m][k] = (k<48 && |k-(m+16)|<=R), serves A and B^T
    __shared__ __attribute__((aligned(32))) _Float16 BND[16 * 64];      //  2,048 B

    int wave = threadIdx.x >> 5;
    int lane = threadIdx.x & 31;

    // init band table (1024 entries / 128 threads)
    for (int e = threadIdx.x; e < 16 * 64; e += 128) {
        int m = e >> 6, k = e & 63;
        int d = k - (m + 16);
        BND[e] = (k < 48 && d <= R && d >= -R) ? (_Float16)1.0f : (_Float16)0.0f;
    }
    __syncthreads();

    int gt = blockIdx.x * 4 + wave;       // global tile id (grid is exact)
    int b  = gt >> 8;
    int t  = gt & 255;
    int ti = t >> 4, tj = t & 15;
    const _Float16* sb = src + (size_t)b * HW;

    // fill transposed halo (zero padded at image edges and rows 48..63)
    for (int e = lane; e < 48 * 64; e += 32) {
        int x = e >> 6, y = e & 63;
        int gy = ti * 16 - 16 + y, gx = tj * 16 - 16 + x;
        _Float16 v = (_Float16)0.0f;
        if (y < 48 && (unsigned)gy < (unsigned)H && (unsigned)gx < (unsigned)W)
            v = sb[gy * W + gx];
        haloT[wave][e] = v;
    }
    // zero T spill padding columns 48..63
    if (lane < 16) {
        #pragma unroll
        for (int k = 48; k < 64; k++) tls[wave][lane * 64 + k] = (_Float16)0.0f;
    }

    int M  = lane & 15;        // A-row / B-col / C-col index for this lane
    int hi = lane >> 4;        // lane-group (0: lanes 0-15, 1: lanes 16-31)

    // row pass: T = A_band(16x48) * X(48x48), three 16-wide column chunks
    #pragma unroll
    for (int cc = 0; cc < 3; cc++) {
        v8f acc = {};
        #pragma unroll
        for (int kb = 0; kb < 64; kb += 32) {
            // A operand: two contiguous 8-half runs of the band table
            v8h alo = *(const v8h*)&BND[M * 64 + kb + hi * 8];
            v8h ahi = *(const v8h*)&BND[M * 64 + kb + 16 + hi * 8];
            v16h a = __builtin_shufflevector(alo, ahi,
                        0,1,2,3,4,5,6,7,8,9,10,11,12,13,14,15);
            // B operand: two contiguous 8-half runs of the transposed halo column
            v8h xlo = *(const v8h*)&haloT[wave][(cc * 16 + M) * 64 + kb + hi * 8];
            v8h xhi = *(const v8h*)&haloT[wave][(cc * 16 + M) * 64 + kb + 16 + hi * 8];
            v16h x = __builtin_shufflevector(xlo, xhi,
                        0,1,2,3,4,5,6,7,8,9,10,11,12,13,14,15);
            acc = __builtin_amdgcn_wmma_f32_16x16x32_f16(
                      false, a, false, x, (short)0, acc, false, false);
        }
        // spill T chunk (C layout -> row-major 16x64 f16 in LDS)
        #pragma unroll
        for (int i = 0; i < 8; i++) {
            int row = i + (hi ? 8 : 0);
            tls[wave][row * 64 + cc * 16 + M] = (_Float16)acc[i];
        }
    }

    // column pass: out = T(16x48) * B_band(48x16)
    v8f out = {};
    #pragma unroll
    for (int kb = 0; kb < 64; kb += 32) {
        v8h alo = *(const v8h*)&tls[wave][M * 64 + kb + hi * 8];
        v8h ahi = *(const v8h*)&tls[wave][M * 64 + kb + 16 + hi * 8];
        v16h a = __builtin_shufflevector(alo, ahi,
                    0,1,2,3,4,5,6,7,8,9,10,11,12,13,14,15);
        v8h blo = *(const v8h*)&BND[M * 64 + kb + hi * 8];
        v8h bhi = *(const v8h*)&BND[M * 64 + kb + 16 + hi * 8];
        v16h bb = __builtin_shufflevector(blo, bhi,
                    0,1,2,3,4,5,6,7,8,9,10,11,12,13,14,15);
        out = __builtin_amdgcn_wmma_f32_16x16x32_f16(
                  false, a, false, bb, (short)0, out, false, false);
    }

    // fused mask-multiply (* src at center) + coalesced store
    float* db = dst + (size_t)b * HW;
    #pragma unroll
    for (int i = 0; i < 8; i++) {
        int row = i + (hi ? 8 : 0);
        float mask = (float)haloT[wave][(16 + M) * 64 + 16 + row];
        db[(ti * 16 + row) * W + tj * 16 + M] = out[i] * mask;
    }
}

// ---------------------------------------------------------------------------
// K4: per-sample argmax (first occurrence) of count
// ---------------------------------------------------------------------------
__global__ void __launch_bounds__(256)
k_argmax(const float* __restrict__ cnt, int* __restrict__ seed,
         float* __restrict__ maxc, int B) {
    __shared__ float sv[256];
    __shared__ int   si[256];
    int b = blockIdx.x;
    const float* c = cnt + (size_t)b * HW;
    float bv = -1.0f; int bi = 0x7FFFFFFF;
    for (int i = threadIdx.x; i < HW; i += 256) {
        float v = c[i];
        if (v > bv) { bv = v; bi = i; }   // stride ascending -> earliest kept
    }
    sv[threadIdx.x] = bv; si[threadIdx.x] = bi;
    __syncthreads();
    for (int s = 128; s > 0; s >>= 1) {
        if (threadIdx.x < s) {
            float v2 = sv[threadIdx.x + s]; int i2 = si[threadIdx.x + s];
            if (v2 > sv[threadIdx.x] ||
                (v2 == sv[threadIdx.x] && i2 < si[threadIdx.x])) {
                sv[threadIdx.x] = v2; si[threadIdx.x] = i2;
            }
        }
        __syncthreads();
    }
    if (threadIdx.x == 0) { seed[b] = si[0]; maxc[b] = sv[0]; }
}

// ---------------------------------------------------------------------------
// K4b: per-image active flag (any sample with positive count)
// ---------------------------------------------------------------------------
__global__ void k_active(const float* __restrict__ maxc, int* __restrict__ act,
                         int nimg) {
    int i = threadIdx.x;
    if (i < nimg) {
        int a = 0;
        for (int s = 0; s < NS; s++) if (maxc[i * NS + s] > 0.0f) a = 1;
        act[i] = a;
    }
}

// ---------------------------------------------------------------------------
// K5: 8-connected flood fill, bit-packed in LDS (8KB region + 8KB filled)
// ---------------------------------------------------------------------------
__global__ void __launch_bounds__(256)
k_flood(const _Float16* __restrict__ fg, _Float16* __restrict__ fill,
        const int* __restrict__ seed, const int* __restrict__ act, int B) {
    __shared__ uint32_t Rg[2048], Fl[2048];
    __shared__ int sch;
    int b = blockIdx.x;
    const _Float16* f = fg + (size_t)b * HW;
    int sidx = seed[b];
    float sval = (float)f[sidx];
    int inv = (sval > 0.5f) ? 0 : 1;       // region = (fg == fg[seed])
    for (int wd = threadIdx.x; wd < 2048; wd += 256) {
        int y = wd >> 3, k = wd & 7;
        uint32_t bits = 0;
        const uint32_t* row = (const uint32_t*)(f + y * W + k * 32);
        #pragma unroll
        for (int i = 0; i < 16; i++) {
            uint32_t w2 = row[i];
            if (w2 & 0xFFFFu) bits |= (1u << (2 * i));
            if (w2 >> 16)     bits |= (1u << (2 * i + 1));
        }
        Rg[wd] = inv ? ~bits : bits;
        Fl[wd] = 0;
    }
    __syncthreads();
    if (threadIdx.x == 0) {
        int sy = sidx / W, sx = sidx - (sidx / W) * W;
        Fl[sy * 8 + (sx >> 5)] |= (1u << (sx & 31));
    }
    __syncthreads();
    for (;;) {
        if (threadIdx.x == 0) sch = 0;
        __syncthreads();
        int local = 0;
        for (int wd = threadIdx.x; wd < 2048; wd += 256) {
            int y = wd >> 3, k = wd & 7;
            uint32_t d = 0;
            #pragma unroll
            for (int dy = -1; dy <= 1; dy++) {
                int yy = y + dy;
                if ((unsigned)yy >= (unsigned)H) continue;
                uint32_t wc = Fl[yy * 8 + k];
                uint32_t wl = (k > 0) ? Fl[yy * 8 + k - 1] : 0u;
                uint32_t wr = (k < 7) ? Fl[yy * 8 + k + 1] : 0u;
                d |= wc | (wc << 1) | (wc >> 1) | (wl >> 31) | (wr << 31);
            }
            uint32_t nf = Fl[wd] | (d & Rg[wd]);
            if (nf != Fl[wd]) { Fl[wd] = nf; local = 1; }
        }
        if (local) atomicOr(&sch, 1);
        __syncthreads();
        int done = (sch == 0);
        __syncthreads();
        if (done) break;
    }
    int a = act[b / NS];
    for (int wd = threadIdx.x; wd < 2048; wd += 256) {
        int y = wd >> 3, k = wd & 7;
        uint32_t bits = a ? Fl[wd] : 0u;
        uint32_t* fo = (uint32_t*)(fill + (size_t)b * HW + y * W + k * 32);
        #pragma unroll
        for (int i = 0; i < 16; i++) {
            uint32_t lo2 = (bits >> (2 * i)) & 1u;
            uint32_t hi2 = (bits >> (2 * i + 1)) & 1u;
            fo[i] = (lo2 ? 0x3C00u : 0u) | (hi2 ? 0x3C000000u : 0u);
        }
    }
}

// ---------------------------------------------------------------------------
// K6: Crew += fg - 2 * ((fc == sc))
// ---------------------------------------------------------------------------
__global__ void k_reward(const _Float16* __restrict__ fg,
                         const float* __restrict__ Fn,
                         const float* __restrict__ Sn,
                         float* __restrict__ Crew, int B) {
    int i = blockIdx.x * blockDim.x + threadIdx.x;
    if (i >= B * HW) return;
    float fgv = (float)fg[i];
    float fn = Fn[i], sn = Sn[i];
    float fc = (fn == 0.0f) ? -1.0f : fn;
    float sc = (sn == 0.0f) ? -2.0f : sn;
    float tmp = (fc == sc) ? 1.0f : 0.0f;
    Crew[i] += fgv - 2.0f * tmp;
}

__global__ void k_zero(float* __restrict__ p, int n) {
    int i = blockIdx.x * blockDim.x + threadIdx.x;
    if (i < n) p[i] = 0.0f;
}

__global__ void k_final(const float* __restrict__ psel,
                        const float* __restrict__ Crew,
                        float* __restrict__ out, int n) {
    int i = blockIdx.x * blockDim.x + threadIdx.x;
    if (i < n) out[i] = psel[i] * Crew[i];
}

// ---------------------------------------------------------------------------
extern "C" void kernel_launch(void* const* d_in, const int* in_sizes, int n_in,
                              void* d_out, int out_size, void* d_ws, size_t ws_size,
                              hipStream_t stream) {
    const float* prob = (const float*)d_in[0];
    int nimg = in_sizes[0] / (NC * HW);    // 8
    int B = nimg * NS;                     // 80
    int tot = B * HW;                      // 5,242,880

    char* ws = (char*)d_ws;
    size_t off = 0;
    auto carve = [&](size_t bytes) -> void* {
        void* p = ws + off;
        off += (bytes + 255) & ~(size_t)255;
        return p;
    };
    uint8_t*  smp  = (uint8_t*)  carve((size_t)tot);
    float*    psel = (float*)    carve((size_t)tot * 4);
    float*    Crew = (float*)    carve((size_t)tot * 4);
    _Float16* fg   = (_Float16*) carve((size_t)tot * 2);
    float*    cnt  = (float*)    carve((size_t)tot * 4);   // count, then Sn
    float*    Fn   = (float*)    carve((size_t)tot * 4);
    _Float16* fill = (_Float16*) carve((size_t)tot * 2);
    int*      seed = (int*)      carve((size_t)B * 4);
    float*    maxc = (float*)    carve((size_t)B * 4);
    int*      act  = (int*)      carve((size_t)nimg * 4);
    (void)ws_size; (void)n_in; (void)out_size;

    int tb = 256;
    k_sample_gather<<<(nimg * HW + tb - 1) / tb, tb, 0, stream>>>(prob, smp, psel, nimg);
    k_zero<<<(tot + tb - 1) / tb, tb, 0, stream>>>(Crew, tot);

    for (int cls = 1; cls < NC; cls++) {
        k_fg<<<(tot + tb - 1) / tb, tb, 0, stream>>>(smp, fg, cls, B);
        k_sumpool_mask_wmma<2><<<B * 64, 128, 0, stream>>>(fg, cnt, B);   // 5x5 count
        k_argmax<<<B, 256, 0, stream>>>(cnt, seed, maxc, B);
        k_active<<<1, 32, 0, stream>>>(maxc, act, nimg);
        k_flood<<<B, 256, 0, stream>>>(fg, fill, seed, act, B);
        k_sumpool_mask_wmma<1><<<B * 64, 128, 0, stream>>>(fill, Fn, B);  // 3x3 Fn
        k_sumpool_mask_wmma<1><<<B * 64, 128, 0, stream>>>(fg, cnt, B);   // 3x3 Sn
        k_reward<<<(tot + tb - 1) / tb, tb, 0, stream>>>(fg, Fn, cnt, Crew, B);
    }

    k_final<<<(tot + tb - 1) / tb, tb, 0, stream>>>(psel, Crew, (float*)d_out, tot);
}
